// Attention_9088150798538
// MI455X (gfx1250) — matrix operand order repaired and checked by reading, not compiled, not run on hardware
//
#include <hip/hip_runtime.h>

// ---------------------------------------------------------------------------
// GQA attention block for MI455X (gfx1250, wave32, WMMA bf16).
// Heavy math via V_WMMA_F32_16X16X32_BF16; tile staging via
// GLOBAL_LOAD_ASYNC_TO_LDS_B128 (ASYNCcnt) where the layout permits.
// ---------------------------------------------------------------------------

typedef __bf16 bf16;
typedef bf16  v16bf __attribute__((ext_vector_type(16)));
typedef bf16  v8bf  __attribute__((ext_vector_type(8)));
typedef float v8f   __attribute__((ext_vector_type(8)));

#define D_MODEL  2048
#define N_HEADS  16
#define N_KV     8
#define HEAD_DIM 128
#define BATCH    4
#define SEQ      2048

#define USE_ASYNC_LDS 1  // flip to 0 if the asm mnemonic is rejected

__device__ __forceinline__ v8bf ld8(const bf16* p) { return *(const v8bf*)p; }

__device__ __forceinline__ v16bf mk16(v8bf lo, v8bf hi) {
    v16bf r;
#pragma unroll
    for (int i = 0; i < 8; ++i) { r[i] = lo[i]; r[i + 8] = hi[i]; }
    return r;
}

// 16-byte global -> LDS copy. Async form writes LDS directly (no VGPR
// round-trip), tracked by ASYNCcnt.
__device__ __forceinline__ void cp16(void* ldsDst, const bf16* gSrc) {
#if USE_ASYNC_LDS
    unsigned l = (unsigned)(uintptr_t)ldsDst;            // addr[31:0] = LDS off
    unsigned long long g = (unsigned long long)(uintptr_t)gSrc;
    asm volatile("global_load_async_to_lds_b128 %0, %1, off"
                 :: "v"(l), "v"(g) : "memory");
#else
    *(v8bf*)ldsDst = *(const v8bf*)gSrc;
#endif
}

__device__ __forceinline__ void cp_wait() {
#if USE_ASYNC_LDS
    asm volatile("s_wait_asynccnt 0x0" ::: "memory");
#endif
}

// ---------------------------------------------------------------------------
// fp32 -> bf16 conversion (grid-strided)
// ---------------------------------------------------------------------------
__global__ void cvt_bf16_kernel(const float* __restrict__ src,
                                bf16* __restrict__ dst, long long n) {
    long long stride = (long long)gridDim.x * blockDim.x;
    for (long long i = (long long)blockIdx.x * blockDim.x + threadIdx.x; i < n;
         i += stride)
        dst[i] = (bf16)src[i];
}

// ---------------------------------------------------------------------------
// RoPE applied in place on bf16 activations.
// ---------------------------------------------------------------------------
__global__ void rope_kernel(bf16* __restrict__ buf, int BT, int T, int nheads,
                            int rowStride) {
    long long total = (long long)BT * nheads * 64;
    long long stride = (long long)gridDim.x * blockDim.x;
    for (long long idx = (long long)blockIdx.x * blockDim.x + threadIdx.x;
         idx < total; idx += stride) {
        int p = (int)(idx & 63);
        long long r = idx >> 6;
        int hh = (int)(r % nheads);
        long long tok = r / nheads;
        int t = (int)(tok % T);
        float inv = __powf(10000.f, ((float)(-2 * p)) / 128.f);
        float ang = (float)t * inv;
        float s = __sinf(ang), c = __cosf(ang);
        size_t base = (size_t)tok * rowStride + (size_t)hh * HEAD_DIM + 2 * p;
        float x1 = (float)buf[base];
        float x2 = (float)buf[base + 1];
        buf[base]     = (bf16)(x1 * c - x2 * s);
        buf[base + 1] = (bf16)(x1 * s + x2 * c);
    }
}

// ---------------------------------------------------------------------------
// WMMA GEMM: out[M,N] = A[M,K](bf16) @ W[N,K]^T (bf16) + bias[N](f32)
// 256 threads / 8 waves, 128x128 tile, K stepped by 64 (2 WMMA sub-steps).
// Wave grid 2x4 -> each wave owns a 64x32 sub-tile = 4x2 C fragments.
// ---------------------------------------------------------------------------
#define KSTEP      64
#define LDS_STRIDE 72  // 64 + 8 pad (bf16 elems)

template <bool OUT_BF16>
__global__ __launch_bounds__(256) void gemm_wmma_kernel(
    const bf16* __restrict__ A, const bf16* __restrict__ W,
    const float* __restrict__ bias, void* __restrict__ out, int M, int N,
    int K) {
    __shared__ __align__(16) bf16 lA[128 * LDS_STRIDE];
    __shared__ __align__(16) bf16 lW[128 * LDS_STRIDE];

    const int tid  = threadIdx.x;
    const int lane = tid & 31;
    const int w    = tid >> 5;
    const int hi   = lane >> 4;
    const int l16  = lane & 15;
    const int waveM = w >> 2;   // 0..1
    const int waveN = w & 3;    // 0..3
    const int tileM = blockIdx.y * 128;
    const int tileN = blockIdx.x * 128;

    v8f acc[4][2];
#pragma unroll
    for (int a = 0; a < 4; ++a)
#pragma unroll
        for (int f = 0; f < 2; ++f)
#pragma unroll
            for (int j = 0; j < 8; ++j) acc[a][f][j] = 0.f;

    for (int k0 = 0; k0 < K; k0 += KSTEP) {
        __syncthreads();
        // Stage A(128x64) and W(128x64) tiles: 1024 16B chunks each, async.
#pragma unroll
        for (int c = 0; c < 4; ++c) {
            int linear = tid + c * 256;
            int row = linear >> 3;       // 8 chunks per 64-wide row
            int col = (linear & 7) * 8;
            cp16(&lA[row * LDS_STRIDE + col],
                 A + (size_t)(tileM + row) * K + k0 + col);
            cp16(&lW[row * LDS_STRIDE + col],
                 W + (size_t)(tileN + row) * K + k0 + col);
        }
        cp_wait();
        __syncthreads();

#pragma unroll
        for (int kk = 0; kk < 2; ++kk) {
            const int kb = kk * 32;
            // Hoist all fragment loads, then batch the 8 WMMAs.
            v16bf af[4], bfr[2];
#pragma unroll
            for (int a = 0; a < 4; ++a) {
                const bf16* pa =
                    &lA[(waveM * 64 + a * 16 + l16) * LDS_STRIDE + kb + 16 * hi];
                af[a] = mk16(ld8(pa), ld8(pa + 8));
            }
#pragma unroll
            for (int f = 0; f < 2; ++f) {
                const bf16* pb =
                    &lW[(waveN * 32 + f * 16 + l16) * LDS_STRIDE + kb +
                        16 * hi];
                bfr[f] = mk16(ld8(pb), ld8(pb + 8));
            }
#pragma unroll
            for (int f = 0; f < 2; ++f)
#pragma unroll
                for (int a = 0; a < 4; ++a)
                    acc[a][f] = __builtin_amdgcn_wmma_f32_16x16x32_bf16(
                        false, af[a], false, bfr[f], (short)0, acc[a][f],
                        false, false);
        }
    }

    // Epilogue: add bias, store.
#pragma unroll
    for (int f = 0; f < 2; ++f) {
        int n = tileN + waveN * 32 + f * 16 + l16;
        float bb = bias[n];
#pragma unroll
        for (int a = 0; a < 4; ++a) {
#pragma unroll
            for (int j = 0; j < 8; ++j) {
                int m = tileM + waveM * 64 + a * 16 + j + 8 * hi;
                float v = acc[a][f][j] + bb;
                if (OUT_BF16)
                    ((bf16*)out)[(size_t)m * N + n] = (bf16)v;
                else
                    ((float*)out)[(size_t)m * N + n] = v;
            }
        }
    }
}

// ---------------------------------------------------------------------------
// Flash attention (causal, GQA): block = (b, h, 64-query tile), 128 threads.
// Wave w owns query rows [64*qt + 16w, +16). K tiles of 64 keys streamed.
// ---------------------------------------------------------------------------
__global__ __launch_bounds__(128) void attn_wmma_kernel(
    const bf16* __restrict__ Qb, const bf16* __restrict__ KVb,
    bf16* __restrict__ Ob, int T) {
    __shared__ __align__(16) bf16 sQ[64 * 128];    // 16 KB
    __shared__ __align__(16) bf16 sK[64 * 128];    // 16 KB
    __shared__ __align__(16) bf16 sVt[128 * 64];   // 16 KB (transposed V)
    __shared__ __align__(16) bf16 sP[4][16 * 64];  //  8 KB (per-wave P)

    const int tid  = threadIdx.x;
    const int lane = tid & 31;
    const int w    = tid >> 5;
    const int hi   = lane >> 4;
    const int l16  = lane & 15;
    const int b   = blockIdx.z;
    const int h   = blockIdx.y;
    const int hkv = h >> 1;  // N_REP = 2
    const int qb  = blockIdx.x * 64;
    const float scale = 0.08838834764831845f;  // 1/sqrt(128)

    // Stage the Q tile (64x128) once, async (waited on in first iteration).
#pragma unroll
    for (int c = 0; c < 8; ++c) {
        int linear = tid + c * 128;
        int row = linear >> 4;
        int col = (linear & 15) * 8;
        cp16(&sQ[row * 128 + col],
             Qb + (size_t)(b * T + qb + row) * D_MODEL + h * HEAD_DIM + col);
    }

    float mstate[8], lstate[8];
    v8f Oacc[8];
#pragma unroll
    for (int j = 0; j < 8; ++j) { mstate[j] = -3.0e38f; lstate[j] = 0.f; }
#pragma unroll
    for (int fd = 0; fd < 8; ++fd)
#pragma unroll
        for (int j = 0; j < 8; ++j) Oacc[fd][j] = 0.f;

    const int nkt = qb / 64 + 1;  // causal: skip fully-masked key tiles
    for (int kt = 0; kt < nkt; ++kt) {
        const int kstart = kt * 64;
        __syncthreads();
        // K tile: async direct-to-LDS. V tile: synchronous (transpose store).
#pragma unroll
        for (int c = 0; c < 8; ++c) {
            int linear = tid + c * 128;
            int row = linear >> 4;
            int col = (linear & 15) * 8;
            size_t gbase =
                (size_t)(b * T + kstart + row) * D_MODEL + hkv * HEAD_DIM + col;
            cp16(&sK[row * 128 + col], KVb + gbase);
            v8bf vv = ld8(KVb + gbase + N_KV * HEAD_DIM);  // V half
#pragma unroll
            for (int i = 0; i < 8; ++i) sVt[(col + i) * 64 + row] = vv[i];
        }
        cp_wait();  // also covers the Q tile on the first iteration
        __syncthreads();
        if (kt + 1 < nkt)  // pull next K/V tile toward L2 (global_prefetch_b8)
            __builtin_prefetch(
                KVb + (size_t)(b * T + kstart + 64) * D_MODEL + hkv * HEAD_DIM,
                0, 1);

        // S = Q * K^T : 16 rows x 64 keys per wave, reduce over d (4x32).
        v8f S[4];
#pragma unroll
        for (int f = 0; f < 4; ++f)
#pragma unroll
            for (int j = 0; j < 8; ++j) S[f][j] = 0.f;
#pragma unroll
        for (int kk = 0; kk < 4; ++kk) {
            const bf16* pa = &sQ[(w * 16 + l16) * 128 + kk * 32 + 16 * hi];
            v16bf a = mk16(ld8(pa), ld8(pa + 8));
            v16bf bb[4];
#pragma unroll
            for (int f = 0; f < 4; ++f) {
                const bf16* pb = &sK[(f * 16 + l16) * 128 + kk * 32 + 16 * hi];
                bb[f] = mk16(ld8(pb), ld8(pb + 8));
            }
#pragma unroll
            for (int f = 0; f < 4; ++f)
                S[f] = __builtin_amdgcn_wmma_f32_16x16x32_bf16(
                    false, a, false, bb[f], (short)0, S[f], false, false);
        }

        // Scale + causal mask (C layout: m = j + 8*hi, n = l16).
#pragma unroll
        for (int f = 0; f < 4; ++f)
#pragma unroll
            for (int j = 0; j < 8; ++j) {
                int qa = qb + w * 16 + j + 8 * hi;
                int na = kstart + f * 16 + l16;
                float s = S[f][j] * scale;
                S[f][j] = (na > qa) ? -3.0e38f : s;
            }

        // Online softmax per row; shfl reductions stay inside 16-lane halves.
#pragma unroll
        for (int j = 0; j < 8; ++j) {
            float mx = -3.0e38f;
#pragma unroll
            for (int f = 0; f < 4; ++f) mx = fmaxf(mx, S[f][j]);
#pragma unroll
            for (int d = 1; d < 16; d <<= 1)
                mx = fmaxf(mx, __shfl_xor(mx, d, 32));
            float newm  = fmaxf(mstate[j], mx);
            float alpha = __expf(mstate[j] - newm);
            mstate[j] = newm;
            float rs = 0.f;
#pragma unroll
            for (int f = 0; f < 4; ++f) {
                float p = __expf(S[f][j] - newm);
                sP[w][(j + 8 * hi) * 64 + f * 16 + l16] = (bf16)p;
                rs += p;
            }
#pragma unroll
            for (int d = 1; d < 16; d <<= 1) rs += __shfl_xor(rs, d, 32);
            lstate[j] = lstate[j] * alpha + rs;
#pragma unroll
            for (int fd = 0; fd < 8; ++fd) Oacc[fd][j] *= alpha;
        }

        // O += P * V  (reduce over 64 keys = 2x32; B fragments from sVt).
#pragma unroll
        for (int kk = 0; kk < 2; ++kk) {
            const bf16* pa = &sP[w][l16 * 64 + kk * 32 + 16 * hi];
            v16bf a = mk16(ld8(pa), ld8(pa + 8));
            v16bf bb[8];
#pragma unroll
            for (int fd = 0; fd < 8; ++fd) {
                const bf16* pb = &sVt[(fd * 16 + l16) * 64 + kk * 32 + 16 * hi];
                bb[fd] = mk16(ld8(pb), ld8(pb + 8));
            }
#pragma unroll
            for (int fd = 0; fd < 8; ++fd)
                Oacc[fd] = __builtin_amdgcn_wmma_f32_16x16x32_bf16(
                    false, a, false, bb[fd], (short)0, Oacc[fd], false, false);
        }
    }

    // Normalize and store O as bf16.
#pragma unroll
    for (int fd = 0; fd < 8; ++fd)
#pragma unroll
        for (int j = 0; j < 8; ++j) {
            int row = qb + w * 16 + j + 8 * hi;
            float v = Oacc[fd][j] / lstate[j];
            Ob[(size_t)(b * T + row) * D_MODEL + h * HEAD_DIM + fd * 16 + l16] =
                (bf16)v;
        }
}

// ---------------------------------------------------------------------------
// Host launcher
// ---------------------------------------------------------------------------
extern "C" void kernel_launch(void* const* d_in, const int* in_sizes, int n_in,
                              void* d_out, int out_size, void* d_ws,
                              size_t ws_size, hipStream_t stream) {
    const float* x   = (const float*)d_in[0];
    const float* Wq  = (const float*)d_in[1];
    const float* bq  = (const float*)d_in[2];
    const float* Wkv = (const float*)d_in[3];
    const float* bkv = (const float*)d_in[4];
    const float* Wo  = (const float*)d_in[5];
    const float* bo  = (const float*)d_in[6];

    const int M = BATCH * SEQ;  // 8192 token rows

    // Workspace carve-out (~152 MB total, all 256B-aligned offsets).
    char* ws = (char*)d_ws;
    bf16* xb   = (bf16*)(ws);
    bf16* Wqb  = (bf16*)(ws + (size_t)33554432);
    bf16* Wkvb = (bf16*)(ws + (size_t)41943040);
    bf16* Wob  = (bf16*)(ws + (size_t)50331648);
    bf16* Qb   = (bf16*)(ws + (size_t)58720256);
    bf16* KVb  = (bf16*)(ws + (size_t)92274688);
    bf16* Ob   = (bf16*)(ws + (size_t)125829120);

    // 1) Downconvert activations + weights to bf16 (bandwidth-trivial).
    cvt_bf16_kernel<<<4096, 256, 0, stream>>>(x, xb, (long long)M * D_MODEL);
    cvt_bf16_kernel<<<2048, 256, 0, stream>>>(Wq, Wqb,
                                              (long long)D_MODEL * D_MODEL);
    cvt_bf16_kernel<<<2048, 256, 0, stream>>>(Wkv, Wkvb,
                                              (long long)D_MODEL * D_MODEL);
    cvt_bf16_kernel<<<2048, 256, 0, stream>>>(Wo, Wob,
                                              (long long)D_MODEL * D_MODEL);

    // 2) QKV projections via WMMA.
    dim3 ggrid(D_MODEL / 128, M / 128);  // (16, 64)
    gemm_wmma_kernel<true><<<ggrid, 256, 0, stream>>>(xb, Wqb, bq, (void*)Qb,
                                                      M, D_MODEL, D_MODEL);
    gemm_wmma_kernel<true><<<ggrid, 256, 0, stream>>>(xb, Wkvb, bkv,
                                                      (void*)KVb, M, D_MODEL,
                                                      D_MODEL);

    // 3) RoPE in-place on Q (16 heads) and K half of KV (8 heads).
    rope_kernel<<<4096, 256, 0, stream>>>(Qb, M, SEQ, N_HEADS, D_MODEL);
    rope_kernel<<<2048, 256, 0, stream>>>(KVb, M, SEQ, N_KV, D_MODEL);

    // 4) Causal GQA flash attention via WMMA.
    dim3 agrid(SEQ / 64, N_HEADS, BATCH);  // (32, 16, 4)
    attn_wmma_kernel<<<agrid, 128, 0, stream>>>(Qb, KVb, Ob, SEQ);

    // 5) Output projection via WMMA, f32 result straight into d_out.
    gemm_wmma_kernel<false><<<ggrid, 256, 0, stream>>>(Ob, Wob, bo, d_out, M,
                                                       D_MODEL, D_MODEL);
}
